// cylinder_fea_37426345017811
// MI455X (gfx1250) — matrix operand order repaired and checked by reading, not compiled, not run on hardware
//
#include <hip/hip_runtime.h>
#include <math.h>

// ---------------- problem constants ----------------
#define NPTS 240000
#define BATCH 2
#define GX 480
#define GY 360
#define GZ 32
#define NCLS 20
#define NVOX (BATCH*GX*GY*GZ)   // 11,059,200
#define EPSBN 1e-5f
#define PI_F 3.14159265358979323846f

// async global->LDS path (gfx1250): guarded so absence falls back to plain copies
#if defined(__has_builtin)
# if __has_builtin(__builtin_amdgcn_global_load_async_to_lds_b128) && __has_builtin(__builtin_amdgcn_s_wait_asynccnt)
#  define USE_ASYNC_LDS 1
# endif
#endif
#ifndef USE_ASYNC_LDS
# define USE_ASYNC_LDS 0
#endif

typedef __attribute__((ext_vector_type(16))) _Float16 v16h;
typedef __attribute__((ext_vector_type(8)))  float    v8f;
typedef __attribute__((ext_vector_type(4)))  unsigned v4u;
typedef __attribute__((ext_vector_type(4)))  int      v4i;

union FragH { v16h h; v4u q[2]; unsigned u[8]; };
union H2    { _Float16 h[2]; unsigned u; };

// monotone float<->uint mapping so unsigned atomicMax == float max
__device__ __forceinline__ unsigned fflip(float f) {
  unsigned u = __float_as_uint(f);
  return (u >> 31) ? ~u : (u | 0x80000000u);
}
__device__ __forceinline__ float funflip(unsigned u) {
  u = (u >> 31) ? (u & 0x7fffffffu) : ~u;
  return __uint_as_float(u);
}

// ---- global->LDS 16B-chunk copy helpers (async when available) ----
__device__ __forceinline__ void g2l_chunk16(void* lds_dst, const void* gsrc) {
#if USE_ASYNC_LDS
  __builtin_amdgcn_global_load_async_to_lds_b128((v4i*)gsrc, (v4i*)lds_dst, 0, 0);
#else
  *(v4u*)lds_dst = *(const v4u*)gsrc;
#endif
}
__device__ __forceinline__ void g2l_copy(void* lds_dst, const void* gsrc, int bytes, int tid) {
  char* d = (char*)lds_dst; const char* s = (const char*)gsrc;
  for (int o = tid * 16; o < bytes; o += 256 * 16) g2l_chunk16(d + o, s + o);
}
__device__ __forceinline__ void g2l_commit() {
#if USE_ASYNC_LDS
  __builtin_amdgcn_s_wait_asynccnt(0);
#endif
}

// ---------------- kernel 1: per-point prep ----------------
__global__ void k_point_prep(const float* __restrict__ pt_fea, const float* __restrict__ xyz,
                             const int* __restrict__ batch_idx, const int* __restrict__ pt_lab,
                             const float* __restrict__ g0, const float* __restrict__ b0,
                             const float* __restrict__ m0, const float* __restrict__ v0,
                             unsigned* __restrict__ counts, unsigned* __restrict__ bits,
                             unsigned* __restrict__ flat, unsigned* __restrict__ act0u,
                             long long* __restrict__ cat_out)
{
  int i = blockIdx.x * 256 + threadIdx.x;
  if (i >= NPTS) return;
  const float minb[3] = {0.f, -PI_F, -4.f};
  const float maxb[3] = {50.f, PI_F, 2.f};
  const float gridf[3] = {(float)GX, (float)GY, (float)GZ};
  int ind[3]; float rel[3];
#pragma unroll
  for (int c = 0; c < 3; ++c) {
    float x   = xyz[(size_t)i * 3 + c];
    float itv = (maxb[c] - minb[c]) / (gridf[c] - 1.f);
    float cl  = fminf(fmaxf(x, minb[c]), maxb[c]);
    int id    = (int)floorf((cl - minb[c]) / itv);
    int hi    = (int)gridf[c] - 1;
    id = id < 0 ? 0 : (id > hi ? hi : id);
    ind[c] = id;
    rel[c] = x - (((float)id + 0.5f) * itv + minb[c]);
  }
  int bi  = batch_idx[i];
  int lab = pt_lab[i];
  unsigned v = ((unsigned)((bi * GX + ind[0]) * GY + ind[1])) * GZ + (unsigned)ind[2];
  flat[i] = v;
  atomicAdd(&counts[(size_t)v * NCLS + lab], 1u);
  atomicOr(&bits[v >> 5], 1u << (v & 31));
  cat_out[(size_t)i * 4 + 0] = (long long)bi;
  cat_out[(size_t)i * 4 + 1] = (long long)ind[0];
  cat_out[(size_t)i * 4 + 2] = (long long)ind[1];
  cat_out[(size_t)i * 4 + 3] = (long long)ind[2];

  float f[9];
  f[0] = rel[0]; f[1] = rel[1]; f[2] = rel[2];
#pragma unroll
  for (int c = 0; c < 6; ++c) f[3 + c] = pt_fea[(size_t)i * 6 + c];
  _Float16 h[10];
#pragma unroll
  for (int c = 0; c < 9; ++c) {
    float s = g0[c] * rsqrtf(v0[c] + EPSBN);
    h[c] = (_Float16)((f[c] - m0[c]) * s + b0[c]);
  }
  h[9] = (_Float16)0.f;
#pragma unroll
  for (int d = 0; d < 16; ++d) {
    H2 p;
    p.h[0] = (d < 5) ? h[2 * d]     : (_Float16)0.f;
    p.h[1] = (d < 5) ? h[2 * d + 1] : (_Float16)0.f;
    act0u[(size_t)i * 16 + d] = p.u;
  }
}

// ---------------- kernel 2: voxel label argmax + occupancy (bitmap-gated) ----------------
__global__ void k_vlab(const unsigned* __restrict__ counts, const unsigned* __restrict__ bits,
                       int* __restrict__ vl, unsigned* __restrict__ occ)
{
  size_t v = (size_t)blockIdx.x * 256 + threadIdx.x;
  if (v >= (size_t)NVOX) return;
  unsigned wbits = bits[v >> 5];
  if (!((wbits >> (v & 31)) & 1u)) { vl[v] = 0; occ[v] = 0u; return; }
  const unsigned* c = &counts[v * NCLS];
  unsigned best = 0; int bl = 0;
#pragma unroll
  for (int k = 0; k < NCLS; ++k) {
    unsigned x = c[k];
    if (x > best) { best = x; bl = k; }   // first max wins (matches argmax)
  }
  vl[v]  = bl;
  occ[v] = 1u;
}

// ---------------- exclusive scan (1024 elems / block) ----------------
__global__ void k_scan(const unsigned* __restrict__ in, unsigned* __restrict__ out,
                       unsigned* __restrict__ sums, int n)
{
  __shared__ unsigned s[256];
  int tid = threadIdx.x;
  size_t base = (size_t)blockIdx.x * 1024 + (size_t)tid * 4;
  unsigned v0 = (base + 0 < (size_t)n) ? in[base + 0] : 0u;
  unsigned v1 = (base + 1 < (size_t)n) ? in[base + 1] : 0u;
  unsigned v2 = (base + 2 < (size_t)n) ? in[base + 2] : 0u;
  unsigned v3 = (base + 3 < (size_t)n) ? in[base + 3] : 0u;
  unsigned tsum = v0 + v1 + v2 + v3;
  s[tid] = tsum;
  __syncthreads();
  for (int off = 1; off < 256; off <<= 1) {
    unsigned t = (tid >= off) ? s[tid - off] : 0u;
    __syncthreads();
    s[tid] += t;
    __syncthreads();
  }
  unsigned excl = s[tid] - tsum;
  if (base + 0 < (size_t)n) out[base + 0] = excl;
  if (base + 1 < (size_t)n) out[base + 1] = excl + v0;
  if (base + 2 < (size_t)n) out[base + 2] = excl + v0 + v1;
  if (base + 3 < (size_t)n) out[base + 3] = excl + v0 + v1 + v2;
  if (sums && tid == 255) sums[blockIdx.x] = s[255];
}

__global__ void k_add(unsigned* __restrict__ out, const unsigned* __restrict__ sums, int n)
{
  size_t i = (size_t)blockIdx.x * 256 + threadIdx.x;
  if (i < (size_t)n) out[i] += sums[i >> 10];
}

// ---------------- unq emit / tail fill / point rank ----------------
__global__ void k_emit(const unsigned* __restrict__ occ, const unsigned* __restrict__ scanEx,
                       long long* __restrict__ unq_out, unsigned* __restrict__ nU)
{
  size_t v = (size_t)blockIdx.x * 256 + threadIdx.x;
  if (v >= (size_t)NVOX) return;
  if (occ[v]) unq_out[scanEx[v]] = (long long)v;
  if (v == (size_t)NVOX - 1) *nU = scanEx[v] + occ[v];
}

__global__ void k_fill(const unsigned* __restrict__ nU, long long* __restrict__ unq_out)
{
  size_t i = (size_t)blockIdx.x * 256 + threadIdx.x;
  if (i >= (size_t)NPTS) return;
  if (i >= (size_t)(*nU)) unq_out[i] = -1ll;
}

__global__ void k_rank(const unsigned* __restrict__ flat, const unsigned* __restrict__ scanEx,
                       unsigned* __restrict__ unq_inv)
{
  size_t i = (size_t)blockIdx.x * 256 + threadIdx.x;
  if (i < (size_t)NPTS) unq_inv[i] = scanEx[flat[i]];
}

// ---------------- parameter pre-pack (once per launch) ----------------
struct MlpParams {
  const float *w1, *b1, *g1, *bb1, *m1, *v1;
  const float *w2, *b2, *g2, *bb2, *m2, *v2;
  const float *w3, *b3, *g3, *bb3, *m3, *v3;
  const float *w4, *b4;
};
struct MlpWs {   // pre-packed f16 weights (transposed [Nc][Kp]) + folded BN scale/shift
  const _Float16 *wt1, *wt2, *wt3, *wt4;
  const float *sc1, *sh1, *sc2, *sh2, *sc3, *sh3, *sh4;
};

__device__ __forceinline__ void cvt_w(_Float16* dst, const float* W, int K, int Kp,
                                      int Nc, int NcFull, int gid, int gs)
{
  for (int idx = gid; idx < Nc * Kp; idx += gs) {
    int n = idx / Kp, k = idx - n * Kp;
    float v = (k < K) ? W[(size_t)k * NcFull + n] : 0.f;
    dst[idx] = (_Float16)v;
  }
}

__global__ void k_prep_params(MlpParams P, const float* __restrict__ wc,
                              _Float16* wt1, _Float16* wt2, _Float16* wt3, _Float16* wt4,
                              _Float16* wtc,
                              float* sc1, float* sh1, float* sc2, float* sh2,
                              float* sc3, float* sh3, float* sh4)
{
  int gid = blockIdx.x * 256 + threadIdx.x;
  int gs  = gridDim.x * 256;
  cvt_w(wt1, P.w1,   9,  32,  64,  64, gid, gs);
  cvt_w(wt2, P.w2,  64,  64, 128, 128, gid, gs);
  cvt_w(wt3, P.w3, 128, 128, 256, 256, gid, gs);
  cvt_w(wt4, P.w4, 256, 256, 256, 256, gid, gs);
  cvt_w(wtc, wc,   256, 256,  16,  16, gid, gs);
  for (int c = gid; c < 64; c += gs) {
    float s = P.g1[c] * rsqrtf(P.v1[c] + EPSBN);
    sc1[c] = s; sh1[c] = (P.b1[c] - P.m1[c]) * s + P.bb1[c];
  }
  for (int c = gid; c < 128; c += gs) {
    float s = P.g2[c] * rsqrtf(P.v2[c] + EPSBN);
    sc2[c] = s; sh2[c] = (P.b2[c] - P.m2[c]) * s + P.bb2[c];
  }
  for (int c = gid; c < 256; c += gs) {
    float s = P.g3[c] * rsqrtf(P.v3[c] + EPSBN);
    sc3[c] = s; sh3[c] = (P.b3[c] - P.m3[c]) * s + P.bb3[c];
  }
  for (int c = gid; c < 256; c += gs) sh4[c] = P.b4[c];
}

// ---------------- fused WMMA MLP ----------------
// one 16-row-per-wave layer: D = A(16xKp f16) x B(Kpx16 f16) tiles, f32 accum
template<int NK, bool RELU, bool POOL>
__device__ __forceinline__ void mlp_layer(const _Float16* __restrict__ actin, int Kp,
                                          _Float16* __restrict__ actout, int wOut,
                                          const _Float16* __restrict__ wt,
                                          const float* __restrict__ sc, const float* __restrict__ sh,
                                          int nTiles, int colBase, int wave, int lane,
                                          size_t rows0, const unsigned* __restrict__ unq_inv,
                                          unsigned* __restrict__ pooled)
{
  const int row = lane & 15;
  const int hs  = lane >> 4;
  FragH af[NK];
  const _Float16* arow = actin + (size_t)(wave * 16 + row) * Kp;
#pragma unroll
  for (int ks = 0; ks < NK; ++ks) {          // A: lanes hold M=row; K halves per ISA layout
    const v4u* p = (const v4u*)(arow + ks * 32 + hs * 8);
    af[ks].q[0] = p[0];                      // K = hs*8 + 0..7
    af[ks].q[1] = p[2];                      // K = 16 + hs*8 + 0..7
  }
  for (int nt = 0; nt < nTiles; ++nt) {
    int nloc = nt * 16 + row;
    const _Float16* wrow = wt + (size_t)nloc * Kp;
    v8f acc = {};
#pragma unroll
    for (int ks = 0; ks < NK; ++ks) {        // B: lane holds col=nloc; K = hs*8 + 0..7, 16 + hs*8 + 0..7
      FragH bf;
      const v4u* q = (const v4u*)(wrow + ks * 32 + hs * 8);
      bf.q[0] = q[0]; bf.q[1] = q[2];
      acc = __builtin_amdgcn_wmma_f32_16x16x32_f16(false, af[ks].h, false, bf.h,
                                                   (short)0, acc, false, false);
    }
    float s = sc[nloc], t = sh[nloc];
#pragma unroll
    for (int r = 0; r < 8; ++r) {            // D: M = hs*8 + r, N = nloc
      float y = acc[r] * s + t;
      if (RELU) y = fmaxf(y, 0.f);
      int M = wave * 16 + hs * 8 + r;
      if (POOL) {
        unsigned rank = unq_inv[rows0 + (size_t)M];
        atomicMax(&pooled[(size_t)rank * 256 + (colBase + nloc)], fflip(y));
      } else {
        actout[(size_t)M * wOut + colBase + nloc] = (_Float16)y;
      }
    }
  }
}

__global__ __launch_bounds__(256) void k_mlp(MlpWs W, const unsigned* __restrict__ act0u,
                                             const int* __restrict__ shuffled,
                                             const unsigned* __restrict__ unq_inv,
                                             unsigned* __restrict__ pooled)
{
  extern __shared__ char smem[];
  _Float16* bufA = (_Float16*)smem;               // 64KB  [128][256] max
  _Float16* bufB = (_Float16*)(smem + 65536);     // 32KB  [128][128] max
  _Float16* wt   = (_Float16*)(smem + 98304);     // 64KB  weight tile
  float*    sc   = (float*)(smem + 163840);       // 1KB
  float*    sh   = (float*)(smem + 164864);       // 1KB  -> total 165888B dyn LDS
  int tid = threadIdx.x, wave = tid >> 5, lane = tid & 31;
  size_t rows0 = (size_t)blockIdx.x * 128;

  // gather act0 rows (feat[shuffled_ind]) -> bufB [128][32] f16 (64B/row, 16B chunks)
  for (int idx = tid; idx < 128 * 4; idx += 256) {
    int lr = idx >> 2, c = idx & 3;
    int src = shuffled[rows0 + (size_t)lr];
    g2l_chunk16((char*)bufB + idx * 16, (const char*)act0u + (size_t)src * 64 + c * 16);
  }
  // L1: 9(->32) -> 64
  g2l_copy(wt, W.wt1, 64 * 32 * 2, tid);
  for (int c = tid; c < 64; c += 256) { sc[c] = W.sc1[c]; sh[c] = W.sh1[c]; }
  g2l_commit();
  __syncthreads();
  mlp_layer<1, true, false>(bufB, 32, bufA, 64, wt, sc, sh, 4, 0, wave, lane, 0, nullptr, nullptr);
  __syncthreads();
  // L2: 64 -> 128
  g2l_copy(wt, W.wt2, 128 * 64 * 2, tid);
  for (int c = tid; c < 128; c += 256) { sc[c] = W.sc2[c]; sh[c] = W.sh2[c]; }
  g2l_commit();
  __syncthreads();
  mlp_layer<2, true, false>(bufA, 64, bufB, 128, wt, sc, sh, 8, 0, wave, lane, 0, nullptr, nullptr);
  __syncthreads();
  // L3: 128 -> 256
  g2l_copy(wt, W.wt3, 256 * 128 * 2, tid);
  for (int c = tid; c < 256; c += 256) { sc[c] = W.sc3[c]; sh[c] = W.sh3[c]; }
  g2l_commit();
  __syncthreads();
  mlp_layer<4, true, false>(bufB, 128, bufA, 256, wt, sc, sh, 16, 0, wave, lane, 0, nullptr, nullptr);
  __syncthreads();
  // L4: 256 -> 256 (no BN/relu), fused segment-max via atomics; two 128-col halves
  g2l_copy(wt, W.wt4, 128 * 256 * 2, tid);
  for (int c = tid; c < 128; c += 256) { sc[c] = 1.f; sh[c] = W.sh4[c]; }
  g2l_commit();
  __syncthreads();
  mlp_layer<8, false, true>(bufA, 256, nullptr, 0, wt, sc, sh, 8, 0, wave, lane, rows0, unq_inv, pooled);
  __syncthreads();
  g2l_copy(wt, W.wt4 + 128 * 256, 128 * 256 * 2, tid);
  for (int c = tid; c < 128; c += 256) { sc[c] = 1.f; sh[c] = W.sh4[128 + c]; }
  g2l_commit();
  __syncthreads();
  mlp_layer<8, false, true>(bufA, 256, nullptr, 0, wt, sc, sh, 8, 128, wave, lane, rows0, unq_inv, pooled);
}

// ---------------- compression: relu(pooled @ wc + bc), WMMA K=256 ----------------
__global__ __launch_bounds__(256) void k_compress(const unsigned* __restrict__ pooled,
                                                  const _Float16* __restrict__ wtc,
                                                  const float* __restrict__ bc,
                                                  const unsigned* __restrict__ nUp,
                                                  float* __restrict__ out)
{
  extern __shared__ char smem[];
  _Float16* xb = (_Float16*)smem;             // 64KB [128][256]
  _Float16* wt = (_Float16*)(smem + 65536);   // 8KB  [16][256] -> total 73728B dyn LDS
  __shared__ float sbc[16];
  int tid = threadIdx.x, wave = tid >> 5, lane = tid & 31;
  unsigned nU = *nUp;
  size_t rows0 = (size_t)blockIdx.x * 128;

  g2l_copy(wt, wtc, 16 * 256 * 2, tid);
  for (int idx = tid; idx < 128 * 128; idx += 256) {   // dwords of xb
    int lr = idx >> 7, d = idx & 127;
    size_t r = rows0 + (size_t)lr;
    H2 p;
    if (r < (size_t)nU) {
      p.h[0] = (_Float16)funflip(pooled[r * 256 + 2 * d]);
      p.h[1] = (_Float16)funflip(pooled[r * 256 + 2 * d + 1]);
    } else { p.h[0] = (_Float16)0.f; p.h[1] = (_Float16)0.f; }  // padded segments -> 0
    ((unsigned*)xb)[idx] = p.u;
  }
  if (tid < 16) sbc[tid] = bc[tid];
  g2l_commit();
  __syncthreads();

  const int row = lane & 15, hs = lane >> 4;
  FragH af[8];
  const _Float16* arow = xb + (size_t)(wave * 16 + row) * 256;
#pragma unroll
  for (int ks = 0; ks < 8; ++ks) {
    const v4u* p = (const v4u*)(arow + ks * 32 + hs * 8);
    af[ks].q[0] = p[0]; af[ks].q[1] = p[2];
  }
  const _Float16* wrow = wt + (size_t)row * 256;
  v8f acc = {};
#pragma unroll
  for (int ks = 0; ks < 8; ++ks) {
    FragH bf;
    const v4u* q = (const v4u*)(wrow + ks * 32 + hs * 8);
    bf.q[0] = q[0]; bf.q[1] = q[2];
    acc = __builtin_amdgcn_wmma_f32_16x16x32_f16(false, af[ks].h, false, bf.h,
                                                 (short)0, acc, false, false);
  }
  float bcv = sbc[row];
#pragma unroll
  for (int r8 = 0; r8 < 8; ++r8) {
    float y = fmaxf(acc[r8] + bcv, 0.f);
    int M = wave * 16 + hs * 8 + r8;
    out[(rows0 + (size_t)M) * 16 + row] = y;
  }
}

// ---------------- host launcher ----------------
extern "C" void kernel_launch(void* const* d_in, const int* in_sizes, int n_in,
                              void* d_out, int out_size, void* d_ws, size_t ws_size,
                              hipStream_t stream)
{
  (void)in_sizes; (void)n_in; (void)out_size; (void)ws_size;
  const float* pt_fea    = (const float*)d_in[0];
  const float* xyz       = (const float*)d_in[1];
  const int*   batch_idx = (const int*)d_in[2];
  const int*   pt_lab    = (const int*)d_in[3];
  const int*   shuffled  = (const int*)d_in[4];
  const float* bn0g = (const float*)d_in[5], *bn0b = (const float*)d_in[6];
  const float* bn0m = (const float*)d_in[7], *bn0v = (const float*)d_in[8];
  MlpParams P;
  P.g1=(const float*)d_in[9];  P.bb1=(const float*)d_in[10]; P.m1=(const float*)d_in[11]; P.v1=(const float*)d_in[12];
  P.g2=(const float*)d_in[13]; P.bb2=(const float*)d_in[14]; P.m2=(const float*)d_in[15]; P.v2=(const float*)d_in[16];
  P.g3=(const float*)d_in[17]; P.bb3=(const float*)d_in[18]; P.m3=(const float*)d_in[19]; P.v3=(const float*)d_in[20];
  P.w1=(const float*)d_in[21]; P.b1=(const float*)d_in[22];
  P.w2=(const float*)d_in[23]; P.b2=(const float*)d_in[24];
  P.w3=(const float*)d_in[25]; P.b3=(const float*)d_in[26];
  P.w4=(const float*)d_in[27]; P.b4=(const float*)d_in[28];
  const float* wc = (const float*)d_in[29];
  const float* bc = (const float*)d_in[30];

  // workspace carve (256B aligned blocks)
  char* w = (char*)d_ws;
  auto take = [&](size_t bytes) -> char* { char* p = w; w += (bytes + 255) & ~(size_t)255; return p; };
  unsigned*  counts  = (unsigned*)take((size_t)NVOX * NCLS * 4);
  unsigned*  bits    = (unsigned*)take((size_t)(NVOX / 32) * 4);
  unsigned*  occ     = (unsigned*)take((size_t)NVOX * 4);
  unsigned*  scanEx  = (unsigned*)take((size_t)NVOX * 4);
  unsigned*  s1      = (unsigned*)take(10800 * 4);
  unsigned*  s1x     = (unsigned*)take(10800 * 4);
  unsigned*  s2      = (unsigned*)take(64);
  unsigned*  s2x     = (unsigned*)take(64);
  unsigned*  nU      = (unsigned*)take(64);
  unsigned*  flat    = (unsigned*)take((size_t)NPTS * 4);
  unsigned*  unq_inv = (unsigned*)take((size_t)NPTS * 4);
  unsigned*  act0    = (unsigned*)take((size_t)NPTS * 16 * 4);   // [N][32] f16
  unsigned*  pooled  = (unsigned*)take((size_t)NPTS * 256 * 4);  // flipped-f32 max
  _Float16*  wt1 = (_Float16*)take(64 * 32 * 2);
  _Float16*  wt2 = (_Float16*)take(128 * 64 * 2);
  _Float16*  wt3 = (_Float16*)take(256 * 128 * 2);
  _Float16*  wt4 = (_Float16*)take(256 * 256 * 2);
  _Float16*  wtc = (_Float16*)take(16 * 256 * 2);
  float* sc1 = (float*)take(64 * 4);  float* sh1 = (float*)take(64 * 4);
  float* sc2 = (float*)take(128 * 4); float* sh2 = (float*)take(128 * 4);
  float* sc3 = (float*)take(256 * 4); float* sh3 = (float*)take(256 * 4);
  float* sh4 = (float*)take(256 * 4);

  MlpWs Wp; Wp.wt1 = wt1; Wp.wt2 = wt2; Wp.wt3 = wt3; Wp.wt4 = wt4;
  Wp.sc1 = sc1; Wp.sh1 = sh1; Wp.sc2 = sc2; Wp.sh2 = sh2;
  Wp.sc3 = sc3; Wp.sh3 = sh3; Wp.sh4 = sh4;

  // outputs packed in return order, native dtypes
  char* ob = (char*)d_out;
  long long* unq_out = (long long*)ob;
  float*     pp_out  = (float*)(ob + (size_t)NPTS * 8);
  int*       vl_out  = (int*)(ob + (size_t)NPTS * 8 + (size_t)NPTS * 16 * 4);
  long long* cat_out = (long long*)(ob + (size_t)NPTS * 8 + (size_t)NPTS * 16 * 4 + (size_t)NVOX * 4);

  (void)hipMemsetAsync(counts, 0, (size_t)NVOX * NCLS * 4, stream);
  (void)hipMemsetAsync(bits, 0, (size_t)(NVOX / 32) * 4, stream);
  (void)hipMemsetAsync(pooled, 0, (size_t)NPTS * 256 * 4, stream);  // 0 == flipped "-inf"

  k_prep_params<<<128, 256, 0, stream>>>(P, wc, wt1, wt2, wt3, wt4, wtc,
                                         sc1, sh1, sc2, sh2, sc3, sh3, sh4);
  k_point_prep<<<(NPTS + 255) / 256, 256, 0, stream>>>(pt_fea, xyz, batch_idx, pt_lab,
                                                       bn0g, bn0b, bn0m, bn0v,
                                                       counts, bits, flat, act0, cat_out);
  k_vlab<<<NVOX / 256, 256, 0, stream>>>(counts, bits, vl_out, occ);
  k_scan<<<NVOX / 1024, 256, 0, stream>>>(occ, scanEx, s1, NVOX);
  k_scan<<<(10800 + 1023) / 1024, 256, 0, stream>>>(s1, s1x, s2, 10800);
  k_scan<<<1, 256, 0, stream>>>(s2, s2x, nullptr, 11);
  k_add<<<(10800 + 255) / 256, 256, 0, stream>>>(s1x, s2x, 10800);
  k_add<<<NVOX / 256, 256, 0, stream>>>(scanEx, s1x, NVOX);
  k_emit<<<NVOX / 256, 256, 0, stream>>>(occ, scanEx, unq_out, nU);
  k_rank<<<(NPTS + 255) / 256, 256, 0, stream>>>(flat, scanEx, unq_inv);
  k_fill<<<(NPTS + 255) / 256, 256, 0, stream>>>(nU, unq_out);
  k_mlp<<<NPTS / 128, 256, 165888, stream>>>(Wp, act0, shuffled, unq_inv, pooled);
  k_compress<<<NPTS / 128, 256, 73728, stream>>>(pooled, wtc, bc, nU, pp_out);
}
